// Discriminator_62208306315772
// MI455X (gfx1250) — matrix operand order repaired and checked by reading, not compiled, not run on hardware
//
#include <hip/hip_runtime.h>

// ---------------- types ----------------
typedef __attribute__((ext_vector_type(16))) _Float16 v16h;
typedef __attribute__((ext_vector_type(8)))  float    v8f;

#define D_IN     64
#define T_STEPS  512
#define BATCH    1024
#define N_TILES  52          // total WMMA B-fragment tiles (GRU weights + w_out)
#define TILE_ELT 512         // 32 lanes * 16 halves per tile (16x16x32 f16 B tile)

// Fragment-tile index bases (must match g_tiles order below)
#define TB_L0_GI 0   // 12 tiles: 6 N-tiles x 2 K-chunks
#define TB_L0_GH 12  // 6
#define TB_L1_GI 18  // 3
#define TB_L1_GH 21  // 3
#define TB_L2_GI 24  // 3 (gate-padded, 8 valid cols each)
#define TB_L2_GH 27  // 3
#define TB_L3_GI 30  // 3
#define TB_L3_GH 33  // 3
#define TB_L4_GI 36  // 6
#define TB_L4_GH 42  // 6
#define TB_OUT   48  // 4

struct TileDesc { unsigned char w, di, cb, nv, kb, kv; };

// B[k][n] = W[cb+n][kb+k]  (B = W^T tile), zero outside (nv, kv)
__constant__ TileDesc g_tiles[N_TILES] = {
    // L0 gi : w_ih_0 [96,64], K chunks 0/32
    {0,64, 0,16, 0,32},{0,64, 0,16,32,32},
    {0,64,16,16, 0,32},{0,64,16,16,32,32},
    {0,64,32,16, 0,32},{0,64,32,16,32,32},
    {0,64,48,16, 0,32},{0,64,48,16,32,32},
    {0,64,64,16, 0,32},{0,64,64,16,32,32},
    {0,64,80,16, 0,32},{0,64,80,16,32,32},
    // L0 gh : w_hh_0 [96,32]
    {1,32, 0,16,0,32},{1,32,16,16,0,32},{1,32,32,16,0,32},
    {1,32,48,16,0,32},{1,32,64,16,0,32},{1,32,80,16,0,32},
    // L1 gi : w_ih_1 [48,32]
    {2,32, 0,16,0,32},{2,32,16,16,0,32},{2,32,32,16,0,32},
    // L1 gh : w_hh_1 [48,16] (K padded)
    {3,16, 0,16,0,16},{3,16,16,16,0,16},{3,16,32,16,0,16},
    // L2 gi : w_ih_2 [24,16], one padded tile per gate (8 valid cols)
    {4,16, 0, 8,0,16},{4,16, 8, 8,0,16},{4,16,16, 8,0,16},
    // L2 gh : w_hh_2 [24,8]
    {5, 8, 0, 8,0, 8},{5, 8, 8, 8,0, 8},{5, 8,16, 8,0, 8},
    // L3 gi : w_ih_3 [48,8]
    {6, 8, 0,16,0, 8},{6, 8,16,16,0, 8},{6, 8,32,16,0, 8},
    // L3 gh : w_hh_3 [48,16]
    {7,16, 0,16,0,16},{7,16,16,16,0,16},{7,16,32,16,0,16},
    // L4 gi : w_ih_4 [96,16]
    {8,16, 0,16,0,16},{8,16,16,16,0,16},{8,16,32,16,0,16},
    {8,16,48,16,0,16},{8,16,64,16,0,16},{8,16,80,16,0,16},
    // L4 gh : w_hh_4 [96,32]
    {9,32, 0,16,0,32},{9,32,16,16,0,32},{9,32,32,16,0,32},
    {9,32,48,16,0,32},{9,32,64,16,0,32},{9,32,80,16,0,32},
    // out : w_out [64,32]
    {10,32, 0,16,0,32},{10,32,16,16,0,32},{10,32,32,16,0,32},{10,32,48,16,0,32},
};

struct WeightPtrs { const float* p[11]; };   // ih0,hh0,ih1,hh1,...,ih4,hh4,w_out

// -------- prep kernel: build f16 B-fragments (WMMA register order) in d_ws --------
// B layout used here (k order matches load_afrag): lane l holds column n=l&15; K = 16*(l>>4) + half_index.
__global__ void build_wfrags_kernel(WeightPtrs wp, _Float16* __restrict__ frag) {
    const int tl = blockIdx.x;
    const int e  = threadIdx.x;          // 0..511
    const TileDesc td = g_tiles[tl];
    const int l  = e >> 4, hi = e & 15;
    const int n  = l & 15;
    const int k  = ((l >> 4) << 4) + hi; // 16*g + hi
    float v = 0.0f;
    if (n < (int)td.nv && k < (int)td.kv)
        v = wp.p[td.w][(td.cb + n) * (int)td.di + (int)td.kb + k];
    frag[tl * TILE_ELT + e] = (_Float16)v;
}

// ---------------- device helpers ----------------
__device__ __forceinline__ v8f splat8(float x) {
    v8f r;
#pragma unroll
    for (int i = 0; i < 8; ++i) r[i] = x;
    return r;
}

__device__ __forceinline__ v8f wmma_acc(v16h a, v16h b, v8f c) {
    return __builtin_amdgcn_wmma_f32_16x16x32_f16(false, a, false, b, (short)0, c,
                                                  false, false);
}

__device__ __forceinline__ v16h load_bfrag(const _Float16* sw, int tile) {
    const _Float16* p = sw + tile * TILE_ELT + (int)threadIdx.x * 16;  // 32B/lane
    v16h r;
    ((uint4*)&r)[0] = *(const uint4*)(p);
    ((uint4*)&r)[1] = *(const uint4*)(p + 8);
    return r;
}

// A-fragment from a row-major 16 x R f16 staging tile (R in halves).
// Lane l: M = l&15; halves 0..15 = K[16g..16g+15].
__device__ __forceinline__ v16h load_afrag(const _Float16* st, int R, int koff) {
    const int l = threadIdx.x;
    const int g = l >> 4, m = l & 15;
    const _Float16* p = st + m * R + koff + g * 16;
    v16h r;
    ((uint4*)&r)[0] = *(const uint4*)(p);
    ((uint4*)&r)[1] = *(const uint4*)(p + 8);
    return r;
}

// Store a 16x16 f32 D tile (C/D layout: VGPR v -> M=v+8g, lane -> N=l&15)
// into a 16x32 f16 row-major staging tile at column base `cb`.
__device__ __forceinline__ void store_htile(_Float16* st, v8f h, int cb) {
    const int l = threadIdx.x;
    const int g = l >> 4;
    _Float16* p = st + (g * 8) * 32 + (l & 15) + cb;
#pragma unroll
    for (int v = 0; v < 8; ++v) p[v * 32] = (_Float16)h[v];
}

// Fast activations: raw v_rcp_f32 / v_exp_f32, no IEEE divide expansion.
__device__ __forceinline__ float sigm_f(float x) {
    return __builtin_amdgcn_rcpf(1.0f + __expf(-x));
}
// tanh(x) = 1 - 2/(1 + e^{2x}); saturates via rcp(inf)=0, no clamp needed.
__device__ __forceinline__ float tanh_f(float x) {
    float e = __expf(2.0f * x);
    return fmaf(-2.0f, __builtin_amdgcn_rcpf(1.0f + e), 1.0f);
}

// torch GRUCell with pre-summed r/z pre-activations:
//   r = sig(pr); z = sig(pz); n = tanh(cn + r*dn); h' = n + z*(h-n)
__device__ __forceinline__ v8f gru2(v8f pr, v8f pz, v8f cn, v8f dn, v8f h) {
    v8f o;
#pragma unroll
    for (int i = 0; i < 8; ++i) {
        float r = sigm_f(pr[i]);
        float z = sigm_f(pz[i]);
        float n = tanh_f(cn[i] + r * dn[i]);
        o[i] = fmaf(z, h[i] - n, n);
    }
    return o;
}

struct GruArgs {
    const float* x;            // [B, T, 64]
    const float* bi[5];        // b_ih_l [3H]
    const float* bh[5];        // b_hh_l [3H]
    const float* b_out;        // [64]
    const _Float16* frags;     // N_TILES * TILE_ELT f16 in d_ws
    float* out;                // [B, T, 64]
};

// -------- main kernel: one wave == 16 batch rows, full 512-step recurrence --------
__global__ __launch_bounds__(32)
__attribute__((amdgpu_waves_per_eu(8)))
void gru_stack_kernel(GruArgs A) {
    __shared__ __align__(32) _Float16 s_w[N_TILES * TILE_ELT]; // 52 KB weights
    __shared__ __align__(16) _Float16 s_x[16 * 64];            // x_t staging (f16)
    __shared__ __align__(16) _Float16 s_h[5][16 * 32];         // h staging, pre-zeroed

    const int lane = threadIdx.x;
    const int b0   = blockIdx.x * 16;
    const int lg   = lane >> 4;
    const int col  = lane & 15;

    // Weights: global (d_ws fragments) -> LDS, 16B chunks
    {
        const uint4* src = (const uint4*)A.frags;
        uint4* dst = (uint4*)s_w;
        const int nvec = N_TILES * TILE_ELT * 2 / 16;
        for (int i = lane; i < nvec; i += 32) dst[i] = src[i];
        unsigned int* z = (unsigned int*)&s_h[0][0];
        for (int i = lane; i < 5 * 16 * 32 / 2; i += 32) z[i] = 0u;
    }
    __syncthreads();

    // Per-lane bias fragments. r/z gates: b_ih+b_hh pre-summed (their WMMA chains
    // accumulate into one register). n gate: kept separate (r*(hn+b_hn) in tanh).
    float brz0[4], bn0[2], bhn0[2];      // L0: r tiles 0..1, z tiles 2..3 ; n tiles 0..1
    float brz1[2], bn1, bhn1;
    float brz2[2], bn2, bhn2;
    float brz3[2], bn3, bhn3;
    float brz4[4], bn4[2], bhn4[2];
    float bo[4];
#pragma unroll
    for (int j = 0; j < 4; ++j) brz0[j] = A.bi[0][16 * j + col] + A.bh[0][16 * j + col];
#pragma unroll
    for (int j = 0; j < 2; ++j) { bn0[j] = A.bi[0][64 + 16 * j + col]; bhn0[j] = A.bh[0][64 + 16 * j + col]; }
#pragma unroll
    for (int j = 0; j < 2; ++j) brz1[j] = A.bi[1][16 * j + col] + A.bh[1][16 * j + col];
    bn1 = A.bi[1][32 + col]; bhn1 = A.bh[1][32 + col];
#pragma unroll
    for (int j = 0; j < 2; ++j)
        brz2[j] = (col < 8) ? (A.bi[2][8 * j + col] + A.bh[2][8 * j + col]) : 0.0f;
    bn2  = (col < 8) ? A.bi[2][16 + col] : 0.0f;
    bhn2 = (col < 8) ? A.bh[2][16 + col] : 0.0f;
#pragma unroll
    for (int j = 0; j < 2; ++j) brz3[j] = A.bi[3][16 * j + col] + A.bh[3][16 * j + col];
    bn3 = A.bi[3][32 + col]; bhn3 = A.bh[3][32 + col];
#pragma unroll
    for (int j = 0; j < 4; ++j) brz4[j] = A.bi[4][16 * j + col] + A.bh[4][16 * j + col];
#pragma unroll
    for (int j = 0; j < 2; ++j) { bn4[j] = A.bi[4][64 + 16 * j + col]; bhn4[j] = A.bh[4][64 + 16 * j + col]; }
#pragma unroll
    for (int j = 0; j < 4; ++j) bo[j] = A.b_out[16 * j + col];

    // Hidden state: f32 D-layout tiles + f16 A-layout fragments (zero init)
    v8f h0a = {}, h0b = {}, h1 = {}, h2 = {}, h3 = {}, h4a = {}, h4b = {};
    v16h ah0 = {}, ah1 = {}, ah2 = {}, ah3 = {}, ah4 = {};

    for (int t = 0; t < T_STEPS; ++t) {
        // ---- x_t [16 rows x 64 f32] -> f16 staging (lane covers half a row) ----
        {
            const int m  = lane >> 1;
            const int hf = (lane & 1) * 32;
            const float* xr = A.x + ((size_t)(b0 + m) * T_STEPS + t) * D_IN + hf;
            _Float16* sp = s_x + m * D_IN + hf;
#pragma unroll
            for (int j = 0; j < 32; j += 4) {
                float4 f = *(const float4*)(xr + j);
                union { _Float16 h[4]; unsigned long long u; } cv;
                cv.h[0] = (_Float16)f.x; cv.h[1] = (_Float16)f.y;
                cv.h[2] = (_Float16)f.z; cv.h[3] = (_Float16)f.w;
                *(unsigned long long*)(sp + j) = cv.u;
            }
            if (t + 1 < T_STEPS) __builtin_prefetch(xr + D_IN, 0, 0);
        }
        __syncthreads();
        v16h ax0 = load_afrag(s_x, 64, 0);
        v16h ax1 = load_afrag(s_x, 64, 32);

        // ---------------- Layer 0 (in 64 -> H 32) ----------------
#pragma unroll
        for (int jj = 0; jj < 2; ++jj) {
            v8f pr = splat8(brz0[jj]);
            pr = wmma_acc(ax0, load_bfrag(s_w, TB_L0_GI + 2 * jj), pr);
            pr = wmma_acc(ax1, load_bfrag(s_w, TB_L0_GI + 2 * jj + 1), pr);
            pr = wmma_acc(ah0, load_bfrag(s_w, TB_L0_GH + jj), pr);
            v8f pz = splat8(brz0[2 + jj]);
            pz = wmma_acc(ax0, load_bfrag(s_w, TB_L0_GI + 4 + 2 * jj), pz);
            pz = wmma_acc(ax1, load_bfrag(s_w, TB_L0_GI + 5 + 2 * jj), pz);
            pz = wmma_acc(ah0, load_bfrag(s_w, TB_L0_GH + 2 + jj), pz);
            v8f cn = splat8(bn0[jj]);
            cn = wmma_acc(ax0, load_bfrag(s_w, TB_L0_GI + 8 + 2 * jj), cn);
            cn = wmma_acc(ax1, load_bfrag(s_w, TB_L0_GI + 9 + 2 * jj), cn);
            v8f dn = wmma_acc(ah0, load_bfrag(s_w, TB_L0_GH + 4 + jj), splat8(bhn0[jj]));
            v8f hn = gru2(pr, pz, cn, dn, jj ? h0b : h0a);
            if (jj) h0b = hn; else h0a = hn;
            store_htile(s_h[0], hn, jj * 16);
        }
        __syncthreads();
        ah0 = load_afrag(s_h[0], 32, 0);

        // ---------------- Layer 1 (32 -> 16) ----------------
        {
            v8f pr = splat8(brz1[0]);
            pr = wmma_acc(ah0, load_bfrag(s_w, TB_L1_GI + 0), pr);
            pr = wmma_acc(ah1, load_bfrag(s_w, TB_L1_GH + 0), pr);
            v8f pz = splat8(brz1[1]);
            pz = wmma_acc(ah0, load_bfrag(s_w, TB_L1_GI + 1), pz);
            pz = wmma_acc(ah1, load_bfrag(s_w, TB_L1_GH + 1), pz);
            v8f cn = wmma_acc(ah0, load_bfrag(s_w, TB_L1_GI + 2), splat8(bn1));
            v8f dn = wmma_acc(ah1, load_bfrag(s_w, TB_L1_GH + 2), splat8(bhn1));
            h1 = gru2(pr, pz, cn, dn, h1);
            store_htile(s_h[1], h1, 0);
        }
        __syncthreads();
        ah1 = load_afrag(s_h[1], 32, 0);

        // ---------------- Layer 2 (16 -> 8, gate-padded tiles) ----------------
        {
            v8f pr = splat8(brz2[0]);
            pr = wmma_acc(ah1, load_bfrag(s_w, TB_L2_GI + 0), pr);
            pr = wmma_acc(ah2, load_bfrag(s_w, TB_L2_GH + 0), pr);
            v8f pz = splat8(brz2[1]);
            pz = wmma_acc(ah1, load_bfrag(s_w, TB_L2_GI + 1), pz);
            pz = wmma_acc(ah2, load_bfrag(s_w, TB_L2_GH + 1), pz);
            v8f cn = wmma_acc(ah1, load_bfrag(s_w, TB_L2_GI + 2), splat8(bn2));
            v8f dn = wmma_acc(ah2, load_bfrag(s_w, TB_L2_GH + 2), splat8(bhn2));
            h2 = gru2(pr, pz, cn, dn, h2);  // cols>=8 stay exactly 0
            store_htile(s_h[2], h2, 0);
        }
        __syncthreads();
        ah2 = load_afrag(s_h[2], 32, 0);

        // ---------------- Layer 3 (8 -> 16) ----------------
        {
            v8f pr = splat8(brz3[0]);
            pr = wmma_acc(ah2, load_bfrag(s_w, TB_L3_GI + 0), pr);
            pr = wmma_acc(ah3, load_bfrag(s_w, TB_L3_GH + 0), pr);
            v8f pz = splat8(brz3[1]);
            pz = wmma_acc(ah2, load_bfrag(s_w, TB_L3_GI + 1), pz);
            pz = wmma_acc(ah3, load_bfrag(s_w, TB_L3_GH + 1), pz);
            v8f cn = wmma_acc(ah2, load_bfrag(s_w, TB_L3_GI + 2), splat8(bn3));
            v8f dn = wmma_acc(ah3, load_bfrag(s_w, TB_L3_GH + 2), splat8(bhn3));
            h3 = gru2(pr, pz, cn, dn, h3);
            store_htile(s_h[3], h3, 0);
        }
        __syncthreads();
        ah3 = load_afrag(s_h[3], 32, 0);

        // ---------------- Layer 4 (16 -> 32) ----------------
#pragma unroll
        for (int jj = 0; jj < 2; ++jj) {
            v8f pr = splat8(brz4[jj]);
            pr = wmma_acc(ah3, load_bfrag(s_w, TB_L4_GI + jj), pr);
            pr = wmma_acc(ah4, load_bfrag(s_w, TB_L4_GH + jj), pr);
            v8f pz = splat8(brz4[2 + jj]);
            pz = wmma_acc(ah3, load_bfrag(s_w, TB_L4_GI + 2 + jj), pz);
            pz = wmma_acc(ah4, load_bfrag(s_w, TB_L4_GH + 2 + jj), pz);
            v8f cn = wmma_acc(ah3, load_bfrag(s_w, TB_L4_GI + 4 + jj), splat8(bn4[jj]));
            v8f dn = wmma_acc(ah4, load_bfrag(s_w, TB_L4_GH + 4 + jj), splat8(bhn4[jj]));
            v8f hn = gru2(pr, pz, cn, dn, jj ? h4b : h4a);
            if (jj) h4b = hn; else h4a = hn;
            store_htile(s_h[4], hn, jj * 16);
        }
        __syncthreads();
        ah4 = load_afrag(s_h[4], 32, 0);

        // ---------------- fused output projection: out_t = h4 @ w_out^T + b_out ----
#pragma unroll
        for (int j = 0; j < 4; ++j) {
            v8f o = wmma_acc(ah4, load_bfrag(s_w, TB_OUT + j), splat8(bo[j]));
            float* op = A.out + ((size_t)(b0 + 8 * lg) * T_STEPS + t) * D_IN + 16 * j + col;
#pragma unroll
            for (int v = 0; v < 8; ++v)
                op[(size_t)v * T_STEPS * D_IN] = o[v];
        }
    }
}

// ---------------- host launch ----------------
extern "C" void kernel_launch(void* const* d_in, const int* in_sizes, int n_in,
                              void* d_out, int out_size, void* d_ws, size_t ws_size,
                              hipStream_t stream) {
    (void)in_sizes; (void)n_in; (void)out_size; (void)ws_size;

    // d_in order: 0 imputed_X; per layer l: 1+4l w_ih, 2+4l w_hh, 3+4l b_ih, 4+4l b_hh;
    //             21 w_out; 22 b_out
    WeightPtrs wp;
    for (int l = 0; l < 5; ++l) {
        wp.p[2 * l]     = (const float*)d_in[1 + 4 * l];
        wp.p[2 * l + 1] = (const float*)d_in[2 + 4 * l];
    }
    wp.p[10] = (const float*)d_in[21];

    _Float16* frags = (_Float16*)d_ws;
    build_wfrags_kernel<<<N_TILES, TILE_ELT, 0, stream>>>(wp, frags);

    GruArgs A;
    A.x = (const float*)d_in[0];
    for (int l = 0; l < 5; ++l) {
        A.bi[l] = (const float*)d_in[3 + 4 * l];
        A.bh[l] = (const float*)d_in[4 + 4 * l];
    }
    A.b_out = (const float*)d_in[22];
    A.frags = frags;
    A.out   = (float*)d_out;

    gru_stack_kernel<<<BATCH / 16, 32, 0, stream>>>(A);
}
